// Simulator_43800076484767
// MI455X (gfx1250) — compile-verified, matrix operand order checked by reading, not run on hardware
//
#include <hip/hip_runtime.h>

// ---------------------------------------------------------------------------
// Types for CDNA5 WMMA (wave32): D(16x16 f32) = A(16x32 bf16) x B(32x16 bf16) + C
// ---------------------------------------------------------------------------
typedef __attribute__((ext_vector_type(16))) __bf16 v16bf;
typedef __attribute__((ext_vector_type(8)))  float  v8f;

union Frag {
    v16bf v;
    uint4 u[2];
};

// cheap bf16 conversion: round-half-up (2 VALU ops)
__device__ __forceinline__ unsigned short f2bf(float x) {
    return (unsigned short)((__float_as_uint(x) + 0x8000u) >> 16);
}

// pack two floats into two bf16 in one dword
__device__ __forceinline__ unsigned int f2bf2(float lo, float hi) {
    unsigned int a = (__float_as_uint(lo) + 0x8000u) >> 16;
    unsigned int b = (__float_as_uint(hi) + 0x8000u) & 0xffff0000u;
    return a | b;
}

__device__ __forceinline__ v8f zerov8() {
    v8f z;
#pragma unroll
    for (int i = 0; i < 8; ++i) z[i] = 0.0f;
    return z;
}

// wave-local LDS RAW fence (CDNA5 split counter); memory clobber stops
// compiler reordering of the ds loads above the ds stores.
__device__ __forceinline__ void lds_fence() {
    asm volatile("s_wait_dscnt 0" ::: "memory");
}

// A-fragment from a row-major bf16 tile in LDS.
// 16-bit A layout (ISA 7.12.2): lane L<16 -> M=L, K = {kb+0..7, kb+16..23}
//                               lane L>=16 -> M=L-16, K = {kb+8..15, kb+24..31}
__device__ __forceinline__ Frag loadA(const unsigned short* tile, int Kpad, int lane, int kbase) {
    int M = lane & 15;
    int g = (lane >> 4) * 8;
    Frag f;
    f.u[0] = *(const uint4*)(tile + M * Kpad + kbase + g);
    f.u[1] = *(const uint4*)(tile + M * Kpad + kbase + 16 + g);
    return f;
}

// B-fragment from pre-swizzled global weights: frag (kc,t) = 32 lanes x 32B.
__device__ __forceinline__ Frag loadB(const uint4* w, int kc, int t, int lane, int NT) {
    int idx = (((kc * NT) + t) * 32 + lane) * 2;
    Frag f;
    f.u[0] = w[idx];
    f.u[1] = w[idx + 1];
    return f;
}

__device__ __forceinline__ v8f wmma_bf16(Frag a, Frag b, v8f c) {
    return __builtin_amdgcn_wmma_f32_16x16x32_bf16(false, a.v, false, b.v, (short)0, c, false, false);
}

// ---------------------------------------------------------------------------
// Weight pre-swizzle: f32 [Kreal][128] -> bf16 B-fragments [kc][t][lane][16]
// ---------------------------------------------------------------------------
__global__ void k_cvtw(const float* __restrict__ W, unsigned short* __restrict__ dst,
                       int Kreal, int Kpad) {
    int nfrag = (Kpad / 32) * 8;
    int tid = blockIdx.x * blockDim.x + threadIdx.x;
    if (tid >= nfrag * 32) return;
    int lane = tid & 31, f = tid >> 5;
    int kc = f >> 3, t = f & 7;
    int g = (lane >> 4) * 8;
    int ncol = t * 16 + (lane & 15);
    union { unsigned short s[16]; uint4 u[2]; } tmp;
#pragma unroll
    for (int i = 0; i < 16; ++i) {
        int k = kc * 32 + ((i < 8) ? 0 : 16) + g + (i & 7);
        float x = (k < Kreal) ? W[k * 128 + ncol] : 0.0f;
        tmp.s[i] = f2bf(x);
    }
    uint4* p = (uint4*)(dst + (size_t)(f * 32 + lane) * 16);
    p[0] = tmp.u[0];
    p[1] = tmp.u[1];
}

// ---------------------------------------------------------------------------
// Small helper kernels
// ---------------------------------------------------------------------------
__global__ void k_gf(const float* dom, const float* t, const float* domn,
                     const float* tn, float* gf) {
    int i = threadIdx.x;
    if (i < 3)       gf[i] = dom[i];
    else if (i < 5)  gf[i] = t[i - 3];
    else if (i < 8)  gf[i] = domn[i - 5];
    else if (i < 10) gf[i] = tn[i - 8];
}

__global__ void k_deg(const int* __restrict__ ei, float* __restrict__ deg, int E_) {
    int e = blockIdx.x * blockDim.x + threadIdx.x;
    if (e < E_) atomicAdd(&deg[ei[E_ + e]], 1.0f);
}

// Embedding MLP: x0(17) -> 128 -> 128, both ReLU. One node per block.
__global__ __launch_bounds__(128) void k_embed(const float* __restrict__ v,
                                               const float* __restrict__ r,
                                               const float* __restrict__ gf,
                                               const float* __restrict__ W1,
                                               const float* __restrict__ b1,
                                               const float* __restrict__ W2,
                                               const float* __restrict__ b2,
                                               float* __restrict__ h, int N_) {
    __shared__ float x0[17];
    __shared__ float hid[128];
    int node = blockIdx.x;
    int j = threadIdx.x;
    if (j == 0)               x0[0] = r[node];
    else if (j < 7)           x0[j] = v[node * 6 + (j - 1)];
    else if (j < 17)          x0[j] = gf[j - 7];
    __syncthreads();
    float acc = b1[j];
    for (int i = 0; i < 17; ++i) acc += x0[i] * W1[i * 128 + j];
    hid[j] = fmaxf(acc, 0.0f);
    __syncthreads();
    float acc2 = b2[j];
    for (int i = 0; i < 128; ++i) acc2 += hid[i] * W2[i * 128 + j];
    h[node * 128 + j] = fmaxf(acc2, 0.0f);
}

// ---------------------------------------------------------------------------
// Edge message kernel: 16 edges per wave, WMMA bf16 MLP, atomic scatter to agg
// ---------------------------------------------------------------------------
#define WPB 4
__global__ __launch_bounds__(128) void k_msg(const float* __restrict__ h,
                                             const float* __restrict__ v,
                                             const float* __restrict__ pos,
                                             const float* __restrict__ r,
                                             const int* __restrict__ ei,
                                             const float* __restrict__ gf,
                                             const uint4* __restrict__ w1,
                                             const uint4* __restrict__ w2,
                                             const float* __restrict__ b1,
                                             const float* __restrict__ b2,
                                             float* __restrict__ agg,
                                             int E_, int ntiles) {
    __shared__ unsigned short smA[WPB][16 * 160];
    __shared__ unsigned short smH[WPB][16 * 128];
    int lane = threadIdx.x & 31;
    int wv = threadIdx.x >> 5;
    int tile = blockIdx.x * WPB + wv;
    if (tile >= ntiles) return;          // uniform per wave
    int e0 = tile * 16;
    unsigned short* A = smA[wv];
    unsigned short* Hs = smH[wv];

    // ---- build m_in tile (16 x 160 bf16): 2 lanes per edge row ----
    {
        int row = lane >> 1, half = lane & 1;
        int e = e0 + row;
        if (e >= E_) e = E_ - 1;
        int s = ei[e], d = ei[E_ + e];
        const float4* hd4 = (const float4*)(h + (size_t)d * 128) + half * 16;
        const float4* hs4 = (const float4*)(h + (size_t)s * 128) + half * 16;
        unsigned short* Ar = A + row * 160 + half * 64;
#pragma unroll 4
        for (int q = 0; q < 16; ++q) {
            float4 a4 = hd4[q];
            float4 b4 = hs4[q];
            uint2 p;
            p.x = f2bf2(a4.x - b4.x, a4.y - b4.y);
            p.y = f2bf2(a4.z - b4.z, a4.w - b4.w);
            *(uint2*)(Ar + q * 4) = p;
        }
        if (half == 0) {
            float sf[32];
            for (int i = 0; i < 6; ++i)  sf[i]      = v[d * 6 + i];
            for (int i = 0; i < 6; ++i)  sf[6 + i]  = v[s * 6 + i];
            sf[12] = r[d];
            sf[13] = r[s];
            for (int i = 0; i < 3; ++i)  sf[14 + i] = pos[d * 3 + i] - pos[s * 3 + i];
            for (int i = 0; i < 10; ++i) sf[17 + i] = gf[i];
            for (int i = 27; i < 32; ++i) sf[i] = 0.0f;
            unsigned short* As = A + row * 160 + 128;
#pragma unroll
            for (int q = 0; q < 8; ++q) {
                uint2 p;
                p.x = f2bf2(sf[q * 4 + 0], sf[q * 4 + 1]);
                p.y = f2bf2(sf[q * 4 + 2], sf[q * 4 + 3]);
                *(uint2*)(As + q * 4) = p;
            }
        }
    }
    lds_fence();

    int ncol = lane & 15;
    int mbase = (lane >> 4) * 8;

    // ---- GEMM1: (16x160) @ (160x128), bias + ReLU -> hidden tile ----
    Frag a1[5];
#pragma unroll
    for (int kc = 0; kc < 5; ++kc) a1[kc] = loadA(A, 160, lane, kc * 32);
    for (int t = 0; t < 8; ++t) {
        v8f c = zerov8();
#pragma unroll
        for (int kc = 0; kc < 5; ++kc) {
            Frag b = loadB(w1, kc, t, lane, 8);
            c = wmma_bf16(a1[kc], b, c);
        }
        int colg = t * 16 + ncol;
        float bias = b1[colg];
#pragma unroll
        for (int i2 = 0; i2 < 8; ++i2) {
            float x = fmaxf(c[i2] + bias, 0.0f);
            Hs[(mbase + i2) * 128 + colg] = f2bf(x);
        }
    }
    lds_fence();

    // ---- GEMM2: (16x128) @ (128x128), bias + ReLU -> atomic scatter ----
    Frag a2[4];
#pragma unroll
    for (int kc = 0; kc < 4; ++kc) a2[kc] = loadA(Hs, 128, lane, kc * 32);
    int drow[8];
#pragma unroll
    for (int i2 = 0; i2 < 8; ++i2) {
        int e = e0 + mbase + i2;
        drow[i2] = ei[E_ + ((e < E_) ? e : (E_ - 1))];
    }
    for (int t = 0; t < 8; ++t) {
        v8f c = zerov8();
#pragma unroll
        for (int kc = 0; kc < 4; ++kc) {
            Frag b = loadB(w2, kc, t, lane, 8);
            c = wmma_bf16(a2[kc], b, c);
        }
        int colg = t * 16 + ncol;
        float bias = b2[colg];
#pragma unroll
        for (int i2 = 0; i2 < 8; ++i2) {
            float x = fmaxf(c[i2] + bias, 0.0f);
            if (e0 + mbase + i2 < E_)
                atomicAdd(&agg[(size_t)drow[i2] * 128 + colg], x);
        }
    }
}

// ---------------------------------------------------------------------------
// Node update kernel: 16 nodes per wave, WMMA bf16 MLP (K=288), residual add
// ---------------------------------------------------------------------------
__global__ __launch_bounds__(128) void k_upd(float* __restrict__ h,
                                             const float* __restrict__ agg,
                                             const float* __restrict__ deg,
                                             const float* __restrict__ gf,
                                             const uint4* __restrict__ w1,
                                             const uint4* __restrict__ w2,
                                             const float* __restrict__ b1,
                                             const float* __restrict__ b2,
                                             int N_, int ntiles) {
    __shared__ unsigned short smA[WPB][16 * 288];
    __shared__ unsigned short smH[WPB][16 * 128];
    int lane = threadIdx.x & 31;
    int wv = threadIdx.x >> 5;
    int tile = blockIdx.x * WPB + wv;
    if (tile >= ntiles) return;          // uniform per wave
    int n0 = tile * 16;
    unsigned short* A = smA[wv];
    unsigned short* Hs = smH[wv];

    // ---- build u_in tile (16 x 288 bf16): [h(128) | agg/deg(128) | gf(10) | pad] ----
    {
        int row = lane >> 1, half = lane & 1;
        int node = n0 + row;
        if (node >= N_) node = N_ - 1;
        unsigned short* Ar = A + row * 288;
        if (half == 0) {
            const float4* hp = (const float4*)(h + (size_t)node * 128);
#pragma unroll 4
            for (int q = 0; q < 32; ++q) {
                float4 a4 = hp[q];
                uint2 p;
                p.x = f2bf2(a4.x, a4.y);
                p.y = f2bf2(a4.z, a4.w);
                *(uint2*)(Ar + q * 4) = p;
            }
        } else {
            float invd = 1.0f / fmaxf(deg[node], 1.0f);
            const float4* gp = (const float4*)(agg + (size_t)node * 128);
#pragma unroll 4
            for (int q = 0; q < 32; ++q) {
                float4 a4 = gp[q];
                uint2 p;
                p.x = f2bf2(a4.x * invd, a4.y * invd);
                p.y = f2bf2(a4.z * invd, a4.w * invd);
                *(uint2*)(Ar + 128 + q * 4) = p;
            }
            float tail[32];
            for (int i = 0; i < 10; ++i) tail[i] = gf[i];
            for (int i = 10; i < 32; ++i) tail[i] = 0.0f;
#pragma unroll
            for (int q = 0; q < 8; ++q) {
                uint2 p;
                p.x = f2bf2(tail[q * 4 + 0], tail[q * 4 + 1]);
                p.y = f2bf2(tail[q * 4 + 2], tail[q * 4 + 3]);
                *(uint2*)(Ar + 256 + q * 4) = p;
            }
        }
    }
    lds_fence();

    int ncol = lane & 15;
    int mbase = (lane >> 4) * 8;

    // ---- GEMM1: (16x288) @ (288x128), bias + ReLU ----
    Frag a1[9];
#pragma unroll
    for (int kc = 0; kc < 9; ++kc) a1[kc] = loadA(A, 288, lane, kc * 32);
    for (int t = 0; t < 8; ++t) {
        v8f c = zerov8();
#pragma unroll
        for (int kc = 0; kc < 9; ++kc) {
            Frag b = loadB(w1, kc, t, lane, 8);
            c = wmma_bf16(a1[kc], b, c);
        }
        int colg = t * 16 + ncol;
        float bias = b1[colg];
#pragma unroll
        for (int i2 = 0; i2 < 8; ++i2) {
            float x = fmaxf(c[i2] + bias, 0.0f);
            Hs[(mbase + i2) * 128 + colg] = f2bf(x);
        }
    }
    lds_fence();

    // ---- GEMM2: (16x128) @ (128x128), bias + ReLU, h += out ----
    Frag a2[4];
#pragma unroll
    for (int kc = 0; kc < 4; ++kc) a2[kc] = loadA(Hs, 128, lane, kc * 32);
    for (int t = 0; t < 8; ++t) {
        v8f c = zerov8();
#pragma unroll
        for (int kc = 0; kc < 4; ++kc) {
            Frag b = loadB(w2, kc, t, lane, 8);
            c = wmma_bf16(a2[kc], b, c);
        }
        int colg = t * 16 + ncol;
        float bias = b2[colg];
#pragma unroll
        for (int i2 = 0; i2 < 8; ++i2) {
            int node = n0 + mbase + i2;
            if (node < N_) {
                float x = fmaxf(c[i2] + bias, 0.0f);
                size_t idx = (size_t)node * 128 + colg;
                h[idx] = h[idx] + x;
            }
        }
    }
}

// ---------------------------------------------------------------------------
// Graph norm (single graph): per-feature sum / sumsq reduction, then apply
// ---------------------------------------------------------------------------
__global__ __launch_bounds__(256) void k_red(const float* __restrict__ h,
                                             float* __restrict__ red, int N_) {
    int f = threadIdx.x & 127;
    int rstart = blockIdx.x * 2 + (threadIdx.x >> 7);
    int stride = gridDim.x * 2;
    float s = 0.0f, ss = 0.0f;
    for (int row = rstart; row < N_; row += stride) {
        float x = h[(size_t)row * 128 + f];
        s += x;
        ss += x * x;
    }
    atomicAdd(&red[f], s);
    atomicAdd(&red[128 + f], ss);
}

__global__ void k_norm(float* __restrict__ h, const float* __restrict__ red,
                       int total, float invn) {
    int i = blockIdx.x * blockDim.x + threadIdx.x;
    if (i >= total) return;
    int f = i & 127;
    float mean = red[f] * invn;
    float var = red[128 + f] * invn - mean * mean;
    float x = h[i];
    h[i] = (x - mean) * rsqrtf(var + 1e-5f);
}

// ---------------------------------------------------------------------------
// Output head: pred = relu(h@W1+b1)@W2 + b2 + [pos, v]; pred_x = mod(domain)
// ---------------------------------------------------------------------------
__global__ __launch_bounds__(128) void k_out(const float* __restrict__ h,
                                             const float* __restrict__ pos,
                                             const float* __restrict__ v,
                                             const float* __restrict__ dnext,
                                             const float* __restrict__ W1,
                                             const float* __restrict__ b1,
                                             const float* __restrict__ W2,
                                             const float* __restrict__ b2,
                                             float* __restrict__ out, int N_) {
    __shared__ float hid[128];
    int node = blockIdx.x;
    int j = threadIdx.x;
    const float* hn = h + (size_t)node * 128;
    float acc = b1[j];
    for (int i = 0; i < 128; ++i) acc += hn[i] * W1[i * 128 + j];
    hid[j] = fmaxf(acc, 0.0f);
    __syncthreads();
    if (j < 9) {
        float acc2 = b2[j];
        for (int i = 0; i < 128; ++i) acc2 += hid[i] * W2[i * 9 + j];
        if (j < 3) {
            float p = acc2 + pos[node * 3 + j];
            float d = dnext[j];
            float q = p - floorf(p / d) * d;     // jnp.remainder
            out[(size_t)node * 3 + j] = q;
        } else {
            float p = acc2 + v[node * 6 + (j - 3)];
            out[(size_t)N_ * 3 + (size_t)node * 6 + (j - 3)] = p;
        }
    }
}

// Macro head from h_mean (uses precomputed per-feature sum in red[0..127])
__global__ __launch_bounds__(128) void k_macro(const float* __restrict__ red,
                                               const float* __restrict__ W1,
                                               const float* __restrict__ b1,
                                               const float* __restrict__ W2,
                                               const float* __restrict__ b2,
                                               float* __restrict__ out,
                                               int N_, float invn) {
    __shared__ float hm[128];
    __shared__ float hid[128];
    int j = threadIdx.x;
    hm[j] = red[j] * invn;
    __syncthreads();
    float acc = b1[j];
    for (int i = 0; i < 128; ++i) acc += hm[i] * W1[i * 128 + j];
    hid[j] = fmaxf(acc, 0.0f);
    __syncthreads();
    if (j < 4) {
        float acc2 = b2[j];
        for (int i = 0; i < 128; ++i) acc2 += hid[i] * W2[i * 4 + j];
        out[(size_t)N_ * 9 + j] = acc2;
    }
}

// ---------------------------------------------------------------------------
// Host launch
// ---------------------------------------------------------------------------
extern "C" void kernel_launch(void* const* d_in, const int* in_sizes, int n_in,
                              void* d_out, int out_size, void* d_ws, size_t ws_size,
                              hipStream_t stream) {
    const float* v           = (const float*)d_in[0];
    const float* pos         = (const float*)d_in[1];
    const float* r           = (const float*)d_in[2];
    const float* domain      = (const float*)d_in[3];
    const float* t           = (const float*)d_in[4];
    const float* domain_next = (const float*)d_in[5];
    const float* t_next      = (const float*)d_in[6];
    const int*   ei          = (const int*)d_in[7];
    const float* emb_W1 = (const float*)d_in[9];
    const float* emb_b1 = (const float*)d_in[10];
    const float* emb_W2 = (const float*)d_in[11];
    const float* emb_b2 = (const float*)d_in[12];
    const float* msg_W1 = (const float*)d_in[13];
    const float* msg_b1 = (const float*)d_in[14];
    const float* msg_W2 = (const float*)d_in[15];
    const float* msg_b2 = (const float*)d_in[16];
    const float* upd_W1 = (const float*)d_in[17];
    const float* upd_b1 = (const float*)d_in[18];
    const float* upd_W2 = (const float*)d_in[19];
    const float* upd_b2 = (const float*)d_in[20];
    const float* out_W1 = (const float*)d_in[21];
    const float* out_b1 = (const float*)d_in[22];
    const float* out_W2 = (const float*)d_in[23];
    const float* out_b2 = (const float*)d_in[24];
    const float* mac_W1 = (const float*)d_in[25];
    const float* mac_b1 = (const float*)d_in[26];
    const float* mac_W2 = (const float*)d_in[27];
    const float* mac_b2 = (const float*)d_in[28];

    int N = in_sizes[0] / 6;
    int E = in_sizes[7] / 2;
    int L = in_sizes[14] / 128;   // msg_b1 is (L,128)

    char* ws = (char*)d_ws;
    size_t offH   = 0;
    size_t offAgg = offH + (size_t)N * 128 * 4;
    size_t offDeg = offAgg + (size_t)N * 128 * 4;
    size_t offW1m = (offDeg + (size_t)N * 4 + 255) & ~(size_t)255;
    size_t offW2m = offW1m + (size_t)160 * 128 * 2;
    size_t offW1u = offW2m + (size_t)128 * 128 * 2;
    size_t offW2u = offW1u + (size_t)288 * 128 * 2;
    size_t offGf  = offW2u + (size_t)128 * 128 * 2;
    size_t offRed = (offGf + 64 + 255) & ~(size_t)255;

    float* h    = (float*)(ws + offH);
    float* agg  = (float*)(ws + offAgg);
    float* deg  = (float*)(ws + offDeg);
    unsigned short* w1m = (unsigned short*)(ws + offW1m);
    unsigned short* w2m = (unsigned short*)(ws + offW2m);
    unsigned short* w1u = (unsigned short*)(ws + offW1u);
    unsigned short* w2u = (unsigned short*)(ws + offW2u);
    float* gfb  = (float*)(ws + offGf);
    float* red  = (float*)(ws + offRed);

    int etiles = (E + 15) / 16;
    int ntilesN = (N + 15) / 16;

    hipMemsetAsync(deg, 0, (size_t)N * 4, stream);
    k_gf<<<1, 16, 0, stream>>>(domain, t, domain_next, t_next, gfb);
    k_deg<<<(E + 255) / 256, 256, 0, stream>>>(ei, deg, E);
    k_embed<<<N, 128, 0, stream>>>(v, r, gfb, emb_W1, emb_b1, emb_W2, emb_b2, h, N);

    for (int l = 0; l < L; ++l) {
        k_cvtw<<<(40 * 32 + 127) / 128, 128, 0, stream>>>(msg_W1 + (size_t)l * 155 * 128, w1m, 155, 160);
        k_cvtw<<<(32 * 32 + 127) / 128, 128, 0, stream>>>(msg_W2 + (size_t)l * 128 * 128, w2m, 128, 128);
        k_cvtw<<<(72 * 32 + 127) / 128, 128, 0, stream>>>(upd_W1 + (size_t)l * 266 * 128, w1u, 266, 288);
        k_cvtw<<<(32 * 32 + 127) / 128, 128, 0, stream>>>(upd_W2 + (size_t)l * 128 * 128, w2u, 128, 128);

        hipMemsetAsync(agg, 0, (size_t)N * 128 * 4, stream);
        k_msg<<<(etiles + WPB - 1) / WPB, 128, 0, stream>>>(
            h, v, pos, r, ei, gfb, (const uint4*)w1m, (const uint4*)w2m,
            msg_b1 + (size_t)l * 128, msg_b2 + (size_t)l * 128, agg, E, etiles);
        k_upd<<<(ntilesN + WPB - 1) / WPB, 128, 0, stream>>>(
            h, agg, deg, gfb, (const uint4*)w1u, (const uint4*)w2u,
            upd_b1 + (size_t)l * 128, upd_b2 + (size_t)l * 128, N, ntilesN);

        hipMemsetAsync(red, 0, 256 * 4, stream);
        k_red<<<512, 256, 0, stream>>>(h, red, N);
        k_norm<<<((N * 128) + 255) / 256, 256, 0, stream>>>(h, red, N * 128, 1.0f / (float)N);
    }

    k_out<<<N, 128, 0, stream>>>(h, pos, v, domain_next, out_W1, out_b1, out_W2, out_b2,
                                 (float*)d_out, N);
    hipMemsetAsync(red, 0, 256 * 4, stream);
    k_red<<<512, 256, 0, stream>>>(h, red, N);
    k_macro<<<1, 128, 0, stream>>>(red, mac_W1, mac_b1, mac_W2, mac_b2,
                                   (float*)d_out, N, 1.0f / (float)N);
}
